// FViTBackbone_21174188770122
// MI455X (gfx1250) — matrix operand order repaired and checked by reading, not compiled, not run on hardware
//
#include <hip/hip_runtime.h>
#include <hip/hip_bf16.h>
#include <math.h>

typedef __attribute__((ext_vector_type(16))) __bf16   bf16x16;
typedef __attribute__((ext_vector_type(8)))  float    f32x8;
typedef __attribute__((ext_vector_type(4)))  unsigned uint4v;
typedef __attribute__((ext_vector_type(8)))  unsigned uint8v;

// ---------------------------------------------------------------- utilities
__device__ inline unsigned short f2bfu(float f) {     // f32 -> bf16 bits, RNE
  unsigned u = __builtin_bit_cast(unsigned, f);
  u += 0x7FFFu + ((u >> 16) & 1u);
  return (unsigned short)(u >> 16);
}
__device__ inline unsigned pack2(float a, float b) {
  return (unsigned)f2bfu(a) | ((unsigned)f2bfu(b) << 16);
}

struct QPair { uint4v q0; uint4v q1; };
__device__ inline bf16x16 frag_lds(const unsigned short* row, int w0, int w1) {
  QPair u;
  u.q0 = *(const uint4v*)(row + 2 * w0);
  u.q1 = *(const uint4v*)(row + 2 * w1);
  return __builtin_bit_cast(bf16x16, u);
}
__device__ inline f32x8 wmma_bf16(bf16x16 a, bf16x16 b, f32x8 c) {
  return __builtin_amdgcn_wmma_f32_16x16x32_bf16(false, a, false, b, (short)0, c, false, false);
}

// ---- TDM: load a 64-row x 32-halfword bf16 tile (row pad 16B -> 80B stride)
// D# per CDNA5 ISA: group0 = {count/mode, lds_addr,
// global_addr, type=2}; group1 = {data_size=2B, pad_enable, pad_interval=16dw,
// pad_amount=4dw, tensor_dim0/1, tile_dim0=32, tile_dim1=64, dim0_stride}.
__device__ inline void tdm_load_b_tile(const unsigned short* gsrc, unsigned lds_byte,
                                       int remK, int remN, long strideElems) {
  unsigned long ga = (unsigned long)(size_t)gsrc;
  uint4v g0;
  g0[0] = 1u;                                             // count=1, user D#
  g0[1] = lds_byte;                                       // LDS byte address
  g0[2] = (unsigned)(ga & 0xffffffffu);                   // global_addr[31:0]
  g0[3] = (unsigned)((ga >> 32) & 0x1ffffffu) | (2u << 30); // [56:32] | type=2
  unsigned t0 = (unsigned)remK, t1 = (unsigned)remN;
  unsigned s0lo = (unsigned)(strideElems & 0xffffffffl);
  unsigned s0hi = (unsigned)((strideElems >> 32) & 0xffffl);
  uint8v g1;
  g1[0] = (1u << 16) | (1u << 20) | (3u << 22) | (3u << 25); // 2B, pad 16dw+4dw
  g1[1] = (t0 << 16);                                     // tensor_dim0[15:0]
  g1[2] = (t0 >> 16) | (t1 << 16);                        // dim0 hi | dim1 lo
  g1[3] = (t1 >> 16) | (32u << 16);                       // dim1 hi | tile_dim0
  g1[4] = 64u;                                            // tile_dim1
  g1[5] = s0lo;                                           // dim0_stride lo
  g1[6] = s0hi;                                           // dim0_stride hi
  g1[7] = 0u;
  uint4v gz = {0u, 0u, 0u, 0u};
  asm volatile("tensor_load_to_lds %0, %1, %2, %3"
               :: "s"(g0), "s"(g1), "s"(gz), "s"(gz)
               : "memory");
}

// ---------------------------------------------------------------- generic WMMA GEMM
struct GemmDesc {
  const float* A; const float* B; const float* bias; const float* resid; float* C;
  const unsigned short* Bbf;   // pre-converted bf16 B panel (N x K row-major) for TDM
  int useTdm;
  int M, N, K;
  int amode; long lda; int aH, aW, aC;
  long ldbn, ldbk;
  int cmode; long ldc; int cH, cW; long obs, ocs; int OW, sy, sx, oy, ox;
  int addbias, dorelu, addresid, beta;
};

__device__ inline long gemm_a_base(const GemmDesc& g, int m) {
  if (g.amode == 0) return (long)m * g.lda;
  int HW = g.aH * g.aW;
  int n = m / HW;
  int p = m - n * HW;
  int off;
  if (g.amode == 1) off = p;
  else { int h = p % g.aH; int w = p / g.aH; off = h * g.aW + w; }
  return (long)n * g.aC * HW + off;
}
__device__ inline void store_one(const GemmDesc& g, long addr, float v) {
  if (g.addresid) v += g.resid[addr];
  if (g.beta)     v += g.C[addr];
  if (g.dorelu && v < 0.f) v = 0.f;
  g.C[addr] = v;
}
__device__ inline void gemm_store_tile(const GemmDesc& g, int mb, int col0, int col1,
                                       const f32x8& a0, const f32x8& a1) {
  const bool c0ok = col0 < g.N, c1ok = col1 < g.N;
  const long cst  = (g.cmode == 0) ? 1 : g.ocs;
  const float bias0 = (g.addbias && c0ok) ? g.bias[col0] : 0.f;
  const float bias1 = (g.addbias && c1ok) ? g.bias[col1] : 0.f;
  const long co0 = (long)col0 * cst, co1 = (long)col1 * cst;
  if (g.cmode == 0) {
#pragma unroll
    for (int r = 0; r < 8; ++r) {
      int m = mb + r;
      if (m < g.M) {
        long rowa = (long)m * g.ldc;
        if (c0ok) store_one(g, rowa + co0, a0[r] + bias0);
        if (c1ok) store_one(g, rowa + co1, a1[r] + bias1);
      }
    }
  } else {
    const int HW = g.cH * g.cW;
    int bidx = mb / HW, p = mb - bidx * HW;
    int h, w;
    if (g.cmode == 1) { h = p / g.cW; w = p - h * g.cW; }
    else              { h = p % g.cH; w = p / g.cH; }
#pragma unroll
    for (int r = 0; r < 8; ++r) {
      if (mb + r < g.M) {
        long rowa = (long)bidx * g.obs + ((long)h * g.sy + g.oy) * g.OW + ((long)w * g.sx + g.ox);
        if (c0ok) store_one(g, rowa + co0, a0[r] + bias0);
        if (c1ok) store_one(g, rowa + co1, a1[r] + bias1);
      }
      if (g.cmode == 1) { if (++w == g.cW) { w = 0; if (++h == g.cH) { h = 0; ++bidx; } } }
      else              { if (++h == g.cH) { h = 0; if (++w == g.cW) { w = 0; ++bidx; } } }
    }
  }
}

#define LSTR 40   // LDS halfword row stride (80B = 16B multiple; TDM pad reproduces it)

__global__ __launch_bounds__(256) void k_wmma_gemm(GemmDesc g) {
  __shared__ unsigned short As[2][64 * LSTR];
  __shared__ unsigned short Bs[2][64 * LSTR];
  const int tid  = threadIdx.x;
  const int lane = tid & 31, wave = tid >> 5;
  const int hi   = lane >> 4, l15 = lane & 15;
  const int m0 = blockIdx.y * 64, n0 = blockIdx.x * 64;
  const int mt = (wave >> 1) * 16, nt = (wave & 1) * 32;

  const int srow = tid >> 2;
  const int skk  = (tid & 3) * 8;
  const int  amr   = m0 + srow;
  const long abase = gemm_a_base(g, amr < g.M ? amr : (g.M - 1));
  const long akst  = (g.amode == 0) ? 1 : (long)(g.aH * g.aW);
  const int  bnr   = n0 + srow;
  const long bbase = (long)(bnr < g.N ? bnr : (g.N - 1)) * g.ldbn;
  const long bkst  = g.ldbk;

  const float* ap = g.A + abase + (long)skk * akst;
  const float* bp = g.B + bbase + (long)skk * bkst;
  const long astep = 32 * akst, bstep = 32 * bkst;
  const int kmain = g.K & ~31;
  const int nk    = kmain >> 5;
  const bool tail = (g.K - kmain) > 0;

  auto stage_main = [&](int buf) {
    unsigned au[4];
#pragma unroll
    for (int e = 0; e < 4; ++e)
      au[e] = pack2(ap[(long)(2 * e) * akst], ap[(long)(2 * e + 1) * akst]);
    *(uint4v*)&As[buf][srow * LSTR + skk] = (uint4v){au[0], au[1], au[2], au[3]};
    ap += astep;
    if (!g.useTdm) {
      unsigned bu[4];
#pragma unroll
      for (int e = 0; e < 4; ++e)
        bu[e] = pack2(bp[(long)(2 * e) * bkst], bp[(long)(2 * e + 1) * bkst]);
      *(uint4v*)&Bs[buf][srow * LSTR + skk] = (uint4v){bu[0], bu[1], bu[2], bu[3]};
      bp += bstep;
    }
  };
  auto stage_tail = [&](int buf) {     // guarded; only for non-TDM shapes (K%32!=0)
    unsigned au[4], bu[4];
#pragma unroll
    for (int e = 0; e < 4; ++e) {
      float av[2], bv[2];
#pragma unroll
      for (int q = 0; q < 2; ++q) {
        int kk = kmain + skk + 2 * e + q;
        int kc = kk < g.K ? kk : (g.K - 1);
        float a = g.A[abase + (long)kc * akst];
        float b = g.B[bbase + (long)kc * bkst];
        av[q] = kk < g.K ? a : 0.f;
        bv[q] = kk < g.K ? b : 0.f;
      }
      au[e] = pack2(av[0], av[1]);
      bu[e] = pack2(bv[0], bv[1]);
    }
    *(uint4v*)&As[buf][srow * LSTR + skk] = (uint4v){au[0], au[1], au[2], au[3]};
    *(uint4v*)&Bs[buf][srow * LSTR + skk] = (uint4v){bu[0], bu[1], bu[2], bu[3]};
  };
  auto tdm_b = [&](int buf, int k0n) {
    tdm_load_b_tile(g.Bbf + (long)n0 * g.K + k0n,
                    (unsigned)(size_t)(void*)&Bs[buf][0],
                    g.K - k0n, g.N - n0, (long)g.K);
  };

  const int aw0 = hi * 8, bw0 = hi * 8;
  f32x8 acc0 = {}, acc1 = {};
  const int total = nk + (tail ? 1 : 0);
  if (nk > 0) {
    if (g.useTdm && wave == 0) tdm_b(0, 0);
    stage_main(0);
  } else if (tail) stage_tail(0);
  int parity = 0;
  for (int it = 0; it < total; ++it) {
    if (g.useTdm) __builtin_amdgcn_s_wait_tensorcnt(0);  // no-op for waves w/ cnt==0
    __syncthreads();                       // buf[parity] staged; old reads done
    int nxt = it + 1;
    if (nxt < nk) {
      __builtin_prefetch(ap, 0, 1);
      if (g.useTdm && wave == 0) tdm_b(parity ^ 1, nxt * 32);
      stage_main(parity ^ 1);
    } else if (nxt == nk && tail) {
      stage_tail(parity ^ 1);
    }
    const unsigned short* arow  = &As[parity][(mt + l15) * LSTR];
    const unsigned short* brow0 = &Bs[parity][(nt + l15) * LSTR];
    const unsigned short* brow1 = &Bs[parity][(nt + 16 + l15) * LSTR];
    bf16x16 af  = frag_lds(arow,  aw0, aw0 + 4);
    bf16x16 bf0 = frag_lds(brow0, bw0, bw0 + 4);
    bf16x16 bf1 = frag_lds(brow1, bw0, bw0 + 4);
    acc0 = wmma_bf16(af, bf0, acc0);
    acc1 = wmma_bf16(af, bf1, acc1);
    parity ^= 1;
  }
  gemm_store_tile(g, m0 + mt + hi * 8, n0 + nt + l15, n0 + nt + 16 + l15, acc0, acc1);
}

// ---------------------------------------------------------------- implicit-GEMM conv (NCHW, OIHW)
struct ConvDesc {
  const float* X; const float* Wt; const float* bias; float* Y;
  int Nb, Cin, IH, IW, stride, pad, OH, OW, O;
};

template <int KK>
__global__ __launch_bounds__(256) void k_wmma_conv(ConvDesc c) {
  __shared__ unsigned short As[2][64 * LSTR];
  __shared__ unsigned short Bs[2][64 * LSTR];
  const int tid  = threadIdx.x;
  const int lane = tid & 31, wave = tid >> 5;
  const int hi   = lane >> 4, l15 = lane & 15;
  const long M   = (long)c.Nb * c.OH * c.OW;
  const int Kdim = c.Cin * KK * KK;
  const int m0 = blockIdx.y * 64, n0 = blockIdx.x * 64;
  const int mt = (wave >> 1) * 16, nt = (wave & 1) * 32;

  const int srow = tid >> 2;
  const int skk  = (tid & 3) * 8;
  const long ar  = (m0 + srow) < M ? (m0 + srow) : (M - 1);
  const bool am_ok = (long)(m0 + srow) < M;
  const int  ow = (int)(ar % c.OW); const long t0 = ar / c.OW;
  const int  oh = (int)(t0 % c.OH); const int nb = (int)(t0 / c.OH);
  const int  ih0 = oh * c.stride - c.pad, iw0 = ow * c.stride - c.pad;
  const long xbase = (long)nb * c.Cin * c.IH * c.IW;
  const long bbase = (long)((n0 + srow) < c.O ? (n0 + srow) : (c.O - 1)) * Kdim;

  const int kmain = Kdim & ~31;
  const int nk    = kmain >> 5;
  const bool tail = (Kdim - kmain) > 0;

  auto stage = [&](int buf, int k0, bool guard) {
    unsigned au[4], bu[4];
#pragma unroll
    for (int e = 0; e < 4; ++e) {
      float av[2], bv[2];
#pragma unroll
      for (int q = 0; q < 2; ++q) {
        int kk = k0 + skk + 2 * e + q;
        int kc = (guard && kk >= Kdim) ? (Kdim - 1) : kk;
        int ci = kc / (KK * KK); int rr = kc - ci * (KK * KK);
        int ky = rr / KK;        int kx = rr - ky * KK;
        int ih = ih0 + ky, iw = iw0 + kx;
        bool inb = am_ok && ih >= 0 && ih < c.IH && iw >= 0 && iw < c.IW &&
                   (!guard || kk < Kdim);
        int ihc = ih < 0 ? 0 : (ih >= c.IH ? c.IH - 1 : ih);
        int iwc = iw < 0 ? 0 : (iw >= c.IW ? c.IW - 1 : iw);
        float a = c.X[xbase + ((long)ci * c.IH + ihc) * c.IW + iwc];
        av[q] = inb ? a : 0.f;
        float b = c.Wt[bbase + kc];
        bv[q] = (!guard || kk < Kdim) ? b : 0.f;
      }
      au[e] = pack2(av[0], av[1]);
      bu[e] = pack2(bv[0], bv[1]);
    }
    *(uint4v*)&As[buf][srow * LSTR + skk] = (uint4v){au[0], au[1], au[2], au[3]};
    *(uint4v*)&Bs[buf][srow * LSTR + skk] = (uint4v){bu[0], bu[1], bu[2], bu[3]};
  };

  const int aw0 = hi * 8, bw0 = hi * 8;
  f32x8 acc0 = {}, acc1 = {};
  const int total = nk + (tail ? 1 : 0);
  if (nk > 0) stage(0, 0, false); else if (tail) stage(0, 0, true);
  int parity = 0;
  for (int it = 0; it < total; ++it) {
    __syncthreads();
    int nxt = it + 1;
    if (nxt < nk)                 stage(parity ^ 1, nxt * 32, false);
    else if (nxt == nk && tail)   stage(parity ^ 1, nxt * 32, true);
    const unsigned short* arow  = &As[parity][(mt + l15) * LSTR];
    const unsigned short* brow0 = &Bs[parity][(nt + l15) * LSTR];
    const unsigned short* brow1 = &Bs[parity][(nt + 16 + l15) * LSTR];
    bf16x16 af  = frag_lds(arow,  aw0, aw0 + 4);
    bf16x16 bf0 = frag_lds(brow0, bw0, bw0 + 4);
    bf16x16 bf1 = frag_lds(brow1, bw0, bw0 + 4);
    acc0 = wmma_bf16(af, bf0, acc0);
    acc1 = wmma_bf16(af, bf1, acc1);
    parity ^= 1;
  }

  const long OHW = (long)c.OH * c.OW;
  long mrow0 = m0 + mt + hi * 8;
  int cc0 = n0 + nt + l15, cc1 = cc0 + 16;
  bool c0ok = cc0 < c.O, c1ok = cc1 < c.O;
  float bias0 = c0ok ? c.bias[cc0] : 0.f;
  float bias1 = c1ok ? c.bias[cc1] : 0.f;
  long co0 = (long)cc0 * OHW, co1 = (long)cc1 * OHW;
  int oww = (int)(mrow0 % c.OW); long tt = mrow0 / c.OW;
  int ohh = (int)(tt % c.OH);    int nbb = (int)(tt / c.OH);
#pragma unroll
  for (int r = 0; r < 8; ++r) {
    if (mrow0 + r < M) {
      long pixbase = (long)nbb * c.O * OHW + (long)ohh * c.OW + oww;
      if (c0ok) c.Y[pixbase + co0] = acc0[r] + bias0;
      if (c1ok) c.Y[pixbase + co1] = acc1[r] + bias1;
    }
    if (++oww == c.OW) { oww = 0; if (++ohh == c.OH) { ohh = 0; ++nbb; } }
  }
}

// ---------------------------------------------------------------- WMMA attention (one (seq, head) per block)
#define PSTR 72
__global__ __launch_bounds__(256) void k_attn_wmma(const float* __restrict__ qkv,
                                                   float* __restrict__ ctx,
                                                   int S, int d, int C, float scale) {
  __shared__ unsigned short Qs[64 * LSTR];
  __shared__ unsigned short Ks[64 * LSTR];
  __shared__ unsigned short Vt[32 * PSTR];
  __shared__ unsigned short Pb[64 * PSTR];
  __shared__ float          P [64 * 64];
  const int b = blockIdx.x, head = blockIdx.y;
  const int tid  = threadIdx.x;
  const int lane = tid & 31, wave = tid >> 5;
  const int hi   = lane >> 4, l15 = lane & 15;
  const long rowbase = (long)b * S;
  const int C3 = 3 * C;

  {
    const int r = tid >> 2, kgrp = (tid & 3) * 8;
    const int rc = r < S ? r : (S - 1);
    const long rbase = (rowbase + rc) * C3 + (long)head * d;
    unsigned qu[4], ku[4];
    unsigned short vv[8];
#pragma unroll
    for (int e = 0; e < 4; ++e) {
      float qf[2], kf[2], vf[2];
#pragma unroll
      for (int q = 0; q < 2; ++q) {
        int kk = kgrp + 2 * e + q;
        int kc = kk < d ? kk : (d - 1);
        bool okv = (r < S) && (kk < d);
        float qq  = qkv[rbase + kc] * scale;
        float kkv = qkv[rbase + kc + C];
        float vvv = qkv[rbase + kc + 2 * C];
        qf[q] = okv ? qq : 0.f; kf[q] = okv ? kkv : 0.f; vf[q] = okv ? vvv : 0.f;
      }
      qu[e] = pack2(qf[0], qf[1]);
      ku[e] = pack2(kf[0], kf[1]);
      vv[2 * e] = f2bfu(vf[0]); vv[2 * e + 1] = f2bfu(vf[1]);
    }
    *(uint4v*)&Qs[r * LSTR + kgrp] = (uint4v){qu[0], qu[1], qu[2], qu[3]};
    *(uint4v*)&Ks[r * LSTR + kgrp] = (uint4v){ku[0], ku[1], ku[2], ku[3]};
#pragma unroll
    for (int e = 0; e < 8; ++e) Vt[(kgrp + e) * PSTR + r] = vv[e];
  }
  for (int i = tid; i < (64 * PSTR) / 2; i += 256) ((unsigned*)Pb)[i] = 0u;
  __syncthreads();

  const int ST = S >> 4;
  for (int t = wave; t < ST * ST; t += 8) {
    int ti = t / ST, tj = t - ti * ST;
    const unsigned short* qr = &Qs[(ti * 16 + l15) * LSTR];
    const unsigned short* kr = &Ks[(tj * 16 + l15) * LSTR];
    bf16x16 af = frag_lds(qr, hi * 8, hi * 8 + 4);
    bf16x16 bf = frag_lds(kr, hi * 8, hi * 8 + 4);
    f32x8 acc = {};
    acc = wmma_bf16(af, bf, acc);
#pragma unroll
    for (int r = 0; r < 8; ++r)
      P[(ti * 16 + r + (hi ? 8 : 0)) * 64 + tj * 16 + l15] = acc[r];
  }
  __syncthreads();

  for (int r = tid; r < S; r += 256) {
    float mx = -1e30f;
    for (int j = 0; j < S; ++j) mx = fmaxf(mx, P[r * 64 + j]);
    float sum = 0.f;
    for (int j = 0; j < S; ++j) { float e = __expf(P[r * 64 + j] - mx); P[r * 64 + j] = e; sum += e; }
    float inv = 1.f / sum;
    for (int j = 0; j < S; ++j) Pb[r * PSTR + j] = f2bfu(P[r * 64 + j] * inv);
  }
  __syncthreads();

  for (int t = wave; t < ST * 2; t += 8) {
    int ti = t >> 1, tc = t & 1;
    const unsigned short* pr = &Pb[(ti * 16 + l15) * PSTR];
    const unsigned short* vr = &Vt[(tc * 16 + l15) * PSTR];
    f32x8 acc = {};
    int kend = (S < 32) ? 32 : S;
    for (int k0 = 0; k0 < kend; k0 += 32) {
      int w = k0 >> 1;
      bf16x16 af = frag_lds(pr, w + hi * 8, w + hi * 8 + 4);
      bf16x16 bf = frag_lds(vr, w + hi * 8, w + hi * 8 + 4);
      acc = wmma_bf16(af, bf, acc);
    }
#pragma unroll
    for (int r = 0; r < 8; ++r) {
      int orow = ti * 16 + r + (hi ? 8 : 0);
      int ocol = tc * 16 + l15;
      if (ocol < d && orow < S)
        ctx[(rowbase + orow) * C + (long)head * d + ocol] = acc[r];
    }
  }
}

// ---------------------------------------------------------------- instance norm
__global__ __launch_bounds__(256) void k_inorm(const float* __restrict__ x, float* __restrict__ y,
                                               int HW, int Csub, int Ctot, int coff, int relu) {
  int plane = blockIdx.x;
  int n = plane / Csub, c = plane % Csub;
  long base = ((long)n * Ctot + coff + c) * HW;
  const float* px = x + base;
  float* py = y + base;
  __shared__ float s1[256], s2[256];
  float a = 0.f, b = 0.f;
  for (int i = threadIdx.x; i < HW; i += 256) { float v = px[i]; a += v; b += v * v; }
  s1[threadIdx.x] = a; s2[threadIdx.x] = b;
  __syncthreads();
  for (int o = 128; o > 0; o >>= 1) {
    if ((int)threadIdx.x < o) { s1[threadIdx.x] += s1[threadIdx.x + o]; s2[threadIdx.x] += s2[threadIdx.x + o]; }
    __syncthreads();
  }
  float m   = s1[0] / HW;
  float var = s2[0] / HW - m * m;
  float inv = rsqrtf(var + 1e-5f);
  for (int i = threadIdx.x; i < HW; i += 256) {
    float v = (px[i] - m) * inv;
    if (relu && v < 0.f) v = 0.f;
    py[i] = v;
  }
}

// ---------------------------------------------------------------- grouped 1x1 MLP
__global__ __launch_bounds__(256) void k_grouped_mlp(const float* __restrict__ x,
    const float* __restrict__ w1, const float* __restrict__ b1,
    const float* __restrict__ w2, const float* __restrict__ b2,
    float* __restrict__ y, long total, int C, int HW) {
  long idx = (long)blockIdx.x * 256 + threadIdx.x;
  if (idx >= total) return;
  long t = idx / HW;
  int c = (int)(t % C);
  float v = x[idx];
  float acc = b2[c] + v;
#pragma unroll
  for (int k = 0; k < 4; ++k) {
    float h = v * w1[c * 4 + k] + b1[c * 4 + k];
    if (h < 0.f) h = 0.f;
    acc += h * w2[c * 4 + k];
  }
  y[idx] = acc;
}

// ---------------------------------------------------------------- depthwise tconv upsample
__global__ __launch_bounds__(256) void k_tconv_dw(const float* __restrict__ x,
    const float* __restrict__ w, const float* __restrict__ b, float* __restrict__ y,
    int Nb, int C, int H, int W, int k, long total) {
  long idx = (long)blockIdx.x * 256 + threadIdx.x;
  if (idx >= total) return;
  int OW = W * k, OH = H * k;
  int ow = (int)(idx % OW); long t = idx / OW;
  int oh = (int)(t % OH); t /= OH;
  int c = (int)(t % C); int n = (int)(t / C);
  int h = oh / k, a = oh % k, wc = ow / k, bb = ow % k;
  y[idx] = x[(((long)n * C + c) * H + h) * W + wc] * w[((long)c * k + a) * k + bb] + b[c];
}

// ---------------------------------------------------------------- f32 -> bf16 panel convert
__global__ __launch_bounds__(256) void k_w2bf(const float* __restrict__ w,
                                              unsigned short* __restrict__ o, long n) {
  long i = (long)blockIdx.x * 256 + threadIdx.x;
  if (i < n) o[i] = f2bfu(w[i]);
}

// ---------------------------------------------------------------- host orchestration
static unsigned short* g_wbf = nullptr;
static void launch_gemm(GemmDesc& g, hipStream_t s) {
  if (g.ldbk == 1 && (g.K & 31) == 0) {     // TDM path: pre-convert weight panel
    long n = (long)g.N * g.K;
    k_w2bf<<<(unsigned)((n + 255) / 256), 256, 0, s>>>(g.B, g_wbf, n);
    g.Bbf = g_wbf; g.useTdm = 1;
  } else { g.Bbf = nullptr; g.useTdm = 0; }
  dim3 grid((unsigned)((g.N + 63) / 64), (unsigned)((g.M + 63) / 64));
  k_wmma_gemm<<<grid, 256, 0, s>>>(g);
}

extern "C" void kernel_launch(void* const* d_in, const int* in_sizes, int n_in,
                              void* d_out, int out_size, void* d_ws, size_t ws_size,
                              hipStream_t stream) {
  (void)in_sizes; (void)n_in; (void)out_size; (void)ws_size;
  const int HS[4]    = {96, 192, 384, 768};
  const int HEADS[3] = {4, 8, 12};

  int ii = 0;
  const float* images = (const float*)d_in[ii++];
  const float *inv_proj_b[3], *inv_proj_w[3], *inv_tconv_b[3], *inv_tconv_w[3];
  for (int s = 0; s < 3; ++s) {
    inv_proj_b[s]  = (const float*)d_in[ii++];
    inv_proj_w[s]  = (const float*)d_in[ii++];
    inv_tconv_b[s] = (const float*)d_in[ii++];
    inv_tconv_w[s] = (const float*)d_in[ii++];
  }
  const float* out_conv_b  = (const float*)d_in[ii++];
  const float* out_conv_w  = (const float*)d_in[ii++];
  const float* out_tconv_b = (const float*)d_in[ii++];
  const float* out_tconv_w = (const float*)d_in[ii++];
  const float* pe_b = (const float*)d_in[ii++];
  const float* pe_w = (const float*)d_in[ii++];
  struct Blk { const float *h_bin,*h_bout,*h_win,*h_wout,*mlp_b1,*mlp_b2,*mlp_w1,*mlp_w2,*v_bin,*v_bout,*v_win,*v_wout; };
  Blk blk[3][2];
  const float *down_b[3], *down_w[3];
  for (int s = 0; s < 3; ++s) {
    for (int bi = 0; bi < 2; ++bi) {
      Blk& B = blk[s][bi];
      B.h_bin  = (const float*)d_in[ii++]; B.h_bout = (const float*)d_in[ii++];
      B.h_win  = (const float*)d_in[ii++]; B.h_wout = (const float*)d_in[ii++];
      B.mlp_b1 = (const float*)d_in[ii++]; B.mlp_b2 = (const float*)d_in[ii++];
      B.mlp_w1 = (const float*)d_in[ii++]; B.mlp_w2 = (const float*)d_in[ii++];
      B.v_bin  = (const float*)d_in[ii++]; B.v_bout = (const float*)d_in[ii++];
      B.v_win  = (const float*)d_in[ii++]; B.v_wout = (const float*)d_in[ii++];
    }
    down_b[s] = (const float*)d_in[ii++];
    down_w[s] = (const float*)d_in[ii++];
  }

  const long ACT = 4l * 96 * 64 * 64;
  float* w = (float*)d_ws;
  float* Xa  = w; w += ACT;
  float* Xb  = w; w += ACT;
  float* XN  = w; w += ACT;
  float* Y   = w; w += ACT;
  float* QKV = w; w += 16384l * 288;
  float* CTX = w; w += 16384l * 96;
  float* UP  = w; w += 4l * 768 * 64 * 64;
  float* FE  = w; w += 4l * 768 * 64 * 64;
  float* BIG = w; w += 4l * 768 * 128 * 128;
  g_wbf = (unsigned short*)w; w += 256l * 1024;   // bf16 weight panel scratch

  {
    ConvDesc c{};
    c.X = images; c.Wt = pe_w; c.bias = pe_b; c.Y = Xa;
    c.Nb = 4; c.Cin = 3; c.IH = 256; c.IW = 256;
    c.stride = 4; c.pad = 0; c.OH = 64; c.OW = 64; c.O = 96;
    long M = 4l * 64 * 64;
    k_wmma_conv<4><<<dim3(2, (unsigned)(M / 64)), 256, 0, stream>>>(c);
    k_inorm<<<4 * 96, 256, 0, stream>>>(Xa, Xa, 64 * 64, 96, 96, 0, 1);
  }

  float* X = Xa; float* XALT = Xb;
  int H = 64, Wd = 64;
  for (int s = 0; s < 3; ++s) {
    int C = HS[s], heads = HEADS[s], dh = C / heads;
    int HW = H * Wd, R = 4 * HW;
    float scale = 1.0f / sqrtf((float)dh);

    for (int bi = 0; bi < 2; ++bi) {
      const Blk& B = blk[s][bi];
      k_inorm<<<4 * C, 256, 0, stream>>>(X, XN, HW, C, C, 0, 0);

      GemmDesc g{};
      g.A = XN; g.B = B.v_win; g.bias = B.v_bin; g.C = QKV;
      g.M = R; g.N = 3 * C; g.K = C;
      g.amode = 1; g.aH = H; g.aW = Wd; g.aC = C;
      g.ldbn = C; g.ldbk = 1;
      g.cmode = 0; g.ldc = 3 * C;
      g.addbias = 1;
      launch_gemm(g, stream);
      k_attn_wmma<<<dim3((unsigned)(R / Wd), (unsigned)heads), 256, 0, stream>>>(QKV, CTX, Wd, dh, C, scale);

      GemmDesc p{};
      p.A = CTX; p.B = B.v_wout; p.bias = B.v_bout; p.resid = XN; p.C = Y;
      p.M = R; p.N = C; p.K = C;
      p.amode = 0; p.lda = C;
      p.ldbn = C; p.ldbk = 1;
      p.cmode = 1; p.cH = H; p.cW = Wd; p.obs = (long)C * HW; p.ocs = HW; p.OW = Wd;
      p.sy = 1; p.sx = 1; p.oy = 0; p.ox = 0;
      p.addbias = 1; p.addresid = 1;
      launch_gemm(p, stream);

      g.B = B.h_win; g.bias = B.h_bin; g.amode = 2;
      launch_gemm(g, stream);
      k_attn_wmma<<<dim3((unsigned)(R / H), (unsigned)heads), 256, 0, stream>>>(QKV, CTX, H, dh, C, scale);
      p.B = B.h_wout; p.bias = B.h_bout; p.cmode = 2; p.addresid = 0; p.resid = nullptr; p.beta = 1;
      launch_gemm(p, stream);

      k_inorm<<<4 * C, 256, 0, stream>>>(Y, XN, HW, C, C, 0, 0);
      long tot = (long)4 * C * HW;
      k_grouped_mlp<<<(unsigned)((tot + 255) / 256), 256, 0, stream>>>(
          XN, B.mlp_w1, B.mlp_b1, B.mlp_w2, B.mlp_b2, X, tot, C, HW);
    }

    int C2 = HS[s + 1], OH = H / 2, OWd = Wd / 2;
    {
      ConvDesc dc{};
      dc.X = X; dc.Wt = down_w[s]; dc.bias = down_b[s]; dc.Y = XALT;
      dc.Nb = 4; dc.Cin = C; dc.IH = H; dc.IW = Wd;
      dc.stride = 2; dc.pad = 1; dc.OH = OH; dc.OW = OWd; dc.O = C2;
      long M = 4l * OH * OWd;
      k_wmma_conv<3><<<dim3((unsigned)((C2 + 63) / 64), (unsigned)((M + 63) / 64)), 256, 0, stream>>>(dc);
    }

    int upk = 2 << s;
    long upTot = 4l * C2 * 64 * 64;
    k_tconv_dw<<<(unsigned)((upTot + 255) / 256), 256, 0, stream>>>(
        XALT, inv_tconv_w[s], inv_tconv_b[s], UP, 4, C2, OH, OWd, upk, upTot);
    k_inorm<<<4 * C2, 256, 0, stream>>>(UP, UP, 64 * 64, C2, C2, 0, 1);
    {
      GemmDesc pr{};
      pr.A = UP; pr.B = inv_proj_w[s]; pr.bias = inv_proj_b[s];
      pr.C = FE + (long)s * 256 * 64 * 64;
      pr.M = 4 * 64 * 64; pr.N = 256; pr.K = C2;
      pr.amode = 1; pr.aH = 64; pr.aW = 64; pr.aC = C2;
      pr.ldbn = C2; pr.ldbk = 1;
      pr.cmode = 1; pr.cH = 64; pr.cW = 64; pr.obs = 768l * 64 * 64; pr.ocs = 64l * 64; pr.OW = 64;
      pr.sy = 1; pr.sx = 1; pr.oy = 0; pr.ox = 0;
      pr.addbias = 1;
      launch_gemm(pr, stream);
    }
    k_inorm<<<4 * 256, 256, 0, stream>>>(FE, FE, 64 * 64, 256, 768, s * 256, 1);

    float* tmp = X; X = XALT; XALT = tmp;
    H = OH; Wd = OWd;
  }

  for (int a = 0; a < 2; ++a)
    for (int b2 = 0; b2 < 2; ++b2) {
      GemmDesc t{};
      t.A = FE; t.B = out_tconv_w + (a * 2 + b2); t.bias = out_tconv_b; t.C = BIG;
      t.M = 4 * 64 * 64; t.N = 768; t.K = 768;
      t.amode = 1; t.aH = 64; t.aW = 64; t.aC = 768;
      t.ldbn = 4; t.ldbk = 768l * 4;
      t.cmode = 1; t.cH = 64; t.cW = 64; t.obs = 768l * 128 * 128; t.ocs = 128l * 128; t.OW = 128;
      t.sy = 2; t.sx = 2; t.oy = a; t.ox = b2;
      t.addbias = 1;
      launch_gemm(t, stream);
    }
  k_inorm<<<4 * 768, 256, 0, stream>>>(BIG, BIG, 128 * 128, 768, 768, 0, 1);

  {
    ConvDesc fc{};
    fc.X = BIG; fc.Wt = out_conv_w; fc.bias = out_conv_b; fc.Y = (float*)d_out;
    fc.Nb = 4; fc.Cin = 768; fc.IH = 128; fc.IW = 128;
    fc.stride = 1; fc.pad = 1; fc.OH = 128; fc.OW = 128; fc.O = 768;
    long M = 4l * 128 * 128;
    k_wmma_conv<3><<<dim3(12, (unsigned)(M / 64)), 256, 0, stream>>>(fc);
  }
}
